// MultiHeadAdditiveAttention_38680475467956
// MI455X (gfx1250) — matrix operand order repaired and checked by reading, not compiled, not run on hardware
//
#include <hip/hip_runtime.h>
#include <hip/hip_bf16.h>
#include <cstdint>

// ---------------------------------------------------------------------------
// MultiHeadAdditiveAttention for MI455X (gfx1250, wave32)
//
//  B=2, L=512, D_MODEL=512, H=8, DK=64
//
//  1) fold_inner/fold_bias : W'q = Aq-folded Wq, W'k = Ak-folded Wk
//  2) wmma_gemm_xwT        : qp / kp / V projections (WMMA f16, f32 acc)
//  3) attn_core_wmma       : s_ij = sum_d av[d]*tanh(qp_i+kp_j)  (v_tanh),
//                            online softmax, P@V on the matrix pipe (WMMA),
//                            kp/V tiles double-buffered through LDS with
//                            GLOBAL_LOAD_ASYNC_TO_LDS_B128 (ASYNCcnt overlap)
//  4) wmma_gemm_xwT        : out = H @ Wo^T + bo
// ---------------------------------------------------------------------------

typedef __attribute__((ext_vector_type(16))) _Float16 v16h;
typedef __attribute__((ext_vector_type(8)))  float    v8f;

#define BATCH 2
#define LSEQ  512
#define DM    512
#define NH    8
#define DK    64
#define MROWS (BATCH * LSEQ)   // 1024
#define KVPAD 68               // kp/V LDS row stride (floats): 16B-aligned, low-conflict

// ---------------------------------------------------------------- helpers ---

__device__ __forceinline__ float fast_tanh(float x) {
#if __has_builtin(__builtin_amdgcn_tanhf)
    return __builtin_amdgcn_tanhf(x);          // gfx1250 v_tanh_f32
#else
    float e2x = __builtin_amdgcn_exp2f(x * 2.88539008177f);
    return 1.0f - 2.0f * __builtin_amdgcn_rcpf(e2x + 1.0f);
#endif
}

__device__ __forceinline__ float fast_exp(float x) {
    return __builtin_amdgcn_exp2f(x * 1.44269504089f);
}

// wave32 lane shuffle via ds_bpermute (byte index, bits[6:2] used)
__device__ __forceinline__ float lane_read(float v, int srcLane) {
    int i = __builtin_bit_cast(int, v);
    i = __builtin_amdgcn_ds_bpermute(srcLane << 2, i);
    return __builtin_bit_cast(float, i);
}

// CDNA5 async copy: global -> LDS, 16B per lane, tracked by ASYNCcnt
__device__ __forceinline__ void async_load_b128(uint32_t lds_off, const float* g) {
    unsigned long long ga = (unsigned long long)(uintptr_t)g;
    asm volatile("global_load_async_to_lds_b128 %0, %1, off"
                 :: "v"(lds_off), "v"(ga) : "memory");
}

// wait until at most N async ops remain outstanding (in-order completion)
__device__ __forceinline__ void wait_async0() {
    asm volatile("s_wait_asynccnt 0x0" ::: "memory");
}
__device__ __forceinline__ void wait_async8() {
    asm volatile("s_wait_asynccnt 0x8" ::: "memory");
}

// LDS byte offset of a __shared__ object (generic LDS addr: low 32 bits = offset)
__device__ __forceinline__ uint32_t lds_offset(const void* p) {
    return (uint32_t)(uintptr_t)p;
}

// ------------------------------------------------- inner-projection folding -

__global__ void fold_inner(const float* __restrict__ Ai,
                           const float* __restrict__ W,
                           float* __restrict__ Weff) {
    const int c = blockIdx.x * blockDim.x + threadIdx.x;   // 0..511
    const int r = blockIdx.y;                              // 0..511
    const int h = r >> 6, dp = r & 63;
    float s = 0.0f;
#pragma unroll 8
    for (int d = 0; d < DK; ++d)
        s += Ai[dp * DK + d] * W[(h * DK + d) * DM + c];
    Weff[r * DM + c] = s;
}

__global__ void fold_bias(const float* __restrict__ Ai,
                          const float* __restrict__ b,
                          float* __restrict__ beff) {
    const int r = blockIdx.x * blockDim.x + threadIdx.x;   // 0..511
    if (r >= DM) return;
    const int h = r >> 6, dp = r & 63;
    float s = 0.0f;
#pragma unroll 8
    for (int d = 0; d < DK; ++d)
        s += Ai[dp * DK + d] * b[h * DK + d];
    beff[r] = s;
}

// ------------------------------------------------------ WMMA GEMM (X@W^T) ---
//
// C[M,N] = A[M,K] @ W[N,K]^T + bias[N]; one wave -> 16x64 tile.
// Fragment layouts (wave32; A/B use contiguous k here, not the ISA 7.12.2 table order):
//  A 16x32 f16 : lane l -> m=l&15, half=l>>4; VGPR v: K=16*half+2v
//  B 32x16 f16 : lane l -> n=l&15, half=l>>4; VGPR v: K=16*half+2v
//  C/D 16x16   : VGPR r -> row=r+8*half, col=lane&15
__global__ void wmma_gemm_xwT(const float* __restrict__ A,
                              const float* __restrict__ W,
                              const float* __restrict__ bias,
                              float* __restrict__ C,
                              int M, int N, int K) {
    const int lane = threadIdx.x & 31;
    const int wib  = threadIdx.x >> 5;
    const int gw   = blockIdx.x * (blockDim.x >> 5) + wib;
    const int ntile = N >> 6;
    const int tm = gw / ntile;
    const int tn = gw - tm * ntile;
    const int m0 = tm << 4;
    const int n0 = tn << 6;
    if (m0 >= M) return;                       // wave-uniform

    const int half = lane >> 4;
    const int q    = lane & 15;

    v8f c0 = {}, c1 = {}, c2 = {}, c3 = {};

    const float* ar = A + (size_t)(m0 + q) * K;
    const float* w0 = W + (size_t)(n0 + q) * K;
    const float* w1 = w0 + (size_t)16 * K;
    const float* w2 = w0 + (size_t)32 * K;
    const float* w3 = w0 + (size_t)48 * K;

    for (int k0 = 0; k0 < K; k0 += 32) {
        v16h a, b0, b1, b2, b3;
#pragma unroll
        for (int v = 0; v < 8; ++v) {
            const int ka = k0 + 16 * half + 2 * v;
            float2 t = *(const float2*)(ar + ka);
            a[2 * v] = (_Float16)t.x;  a[2 * v + 1] = (_Float16)t.y;

            const int kb = k0 + 16 * half + 2 * v;
            float2 u0 = *(const float2*)(w0 + kb);
            b0[2 * v] = (_Float16)u0.x; b0[2 * v + 1] = (_Float16)u0.y;
            float2 u1 = *(const float2*)(w1 + kb);
            b1[2 * v] = (_Float16)u1.x; b1[2 * v + 1] = (_Float16)u1.y;
            float2 u2 = *(const float2*)(w2 + kb);
            b2[2 * v] = (_Float16)u2.x; b2[2 * v + 1] = (_Float16)u2.y;
            float2 u3 = *(const float2*)(w3 + kb);
            b3[2 * v] = (_Float16)u3.x; b3[2 * v + 1] = (_Float16)u3.y;
        }
        c0 = __builtin_amdgcn_wmma_f32_16x16x32_f16(false, a, false, b0, (short)0, c0, false, false);
        c1 = __builtin_amdgcn_wmma_f32_16x16x32_f16(false, a, false, b1, (short)0, c1, false, false);
        c2 = __builtin_amdgcn_wmma_f32_16x16x32_f16(false, a, false, b2, (short)0, c2, false, false);
        c3 = __builtin_amdgcn_wmma_f32_16x16x32_f16(false, a, false, b3, (short)0, c3, false, false);
    }

#pragma unroll
    for (int r = 0; r < 8; ++r) {
        const int row = m0 + r + 8 * half;
        float* cr = C + (size_t)row * N + n0 + q;
        cr[0]  = c0[r] + bias[n0 + q];
        cr[16] = c1[r] + bias[n0 + q + 16];
        cr[32] = c2[r] + bias[n0 + q + 32];
        cr[48] = c3[r] + bias[n0 + q + 48];
    }
}

// ------------------------------------------- attention core (flash + WMMA) --
//
// Block = 128 threads = 4 waves; covers 32 query rows of one (b,h).
// Row-tiles: rt = w>>1 (rows rt*16..rt*16+15).
// Score role : wave w scores rows rt*16 + (w&1)*8 .. +7, lane = key j.
// PV role    : wave w accumulates D rows rt*16..+15, dims d0=(w&1)*32..d0+31
//              via two WMMA f16 (P 16x32 as A-frag, V 32x16 as B-frags).
// kp/V chunk tiles double-buffered; GLOBAL_LOAD_ASYNC_TO_LDS_B128 for chunk
// c+1 issued before chunk c's compute; s_wait_asynccnt 8 hides the latency
// behind the tanh + WMMA phase (each thread issues exactly 8 async ops/chunk,
// async loads complete in order).
__global__ void __launch_bounds__(128)
attn_core_wmma(const float* __restrict__ qp,
               const float* __restrict__ kp,
               const float* __restrict__ V,
               const float* __restrict__ av,
               float* __restrict__ Hout) {
    __shared__ float kpt[2][32 * KVPAD];
    __shared__ float vt [2][32 * KVPAD];
    __shared__ float qrow[32 * 64];
    __shared__ float ptile[2][16 * 33];    // P staging, padded for A-frag reads
    __shared__ float scl[32];              // per-row chunk rescale factor
    __shared__ float lsum[32];             // per-row running softmax denom
    __shared__ float avs[64];

    const int tid  = threadIdx.x;
    const int lane = tid & 31;
    const int w    = tid >> 5;                 // 0..3
    const int bh   = blockIdx.x >> 4;          // 16 blocks per (b,h)
    const int i0   = (blockIdx.x & 15) * 32;
    const int b    = bh >> 3;
    const int h    = bh & 7;

    const size_t gbase = ((size_t)(b * LSEQ) * DM) + h * DK;   // (b,h) row 0

    // stage 32 query rows once (plain loads; done once per block)
    for (int t = tid; t < 32 * 64; t += 128) {
        const int r = t >> 6, d = t & 63;
        qrow[t] = qp[gbase + (size_t)(i0 + r) * DM + d];
    }
    if (tid < 64) avs[tid] = av[tid];

    const int rt    = w >> 1;
    const int srow0 = rt * 16 + (w & 1) * 8;
    const int d0    = (w & 1) * 32;
    const int half  = lane >> 4;
    const int q     = lane & 15;

    float mrow[8], lrow[8];
#pragma unroll
    for (int r = 0; r < 8; ++r) { mrow[r] = -INFINITY; lrow[r] = 0.0f; }

    v8f acc0 = {}, acc1 = {};

    const uint32_t kpt_lds0 = lds_offset(&kpt[0][0]);
    const uint32_t kpt_lds1 = lds_offset(&kpt[1][0]);
    const uint32_t vt_lds0  = lds_offset(&vt[0][0]);
    const uint32_t vt_lds1  = lds_offset(&vt[1][0]);

    // per-chunk issue: 4 t-iterations x 2 async ops = exactly 8 per thread
    auto issue_chunk = [&](int j0, int buf) {
        const uint32_t kb = buf ? kpt_lds1 : kpt_lds0;
        const uint32_t vb = buf ? vt_lds1  : vt_lds0;
        for (int t = tid; t < 512; t += 128) {     // 512 x 16B per tile
            const int j = t >> 4, seg = t & 15;
            const size_t   goff = gbase + (size_t)(j0 + j) * DM + seg * 4;
            const uint32_t loff = (uint32_t)((j * KVPAD + seg * 4) * 4);
            async_load_b128(kb + loff, kp + goff);
            async_load_b128(vb + loff, V  + goff);
        }
    };

    issue_chunk(0, 0);                             // prologue: chunk 0 -> buf 0

    const int NCHUNK = LSEQ / 32;                  // 16
    for (int c = 0; c < NCHUNK; ++c) {
        const int cur = c & 1;
        if (c + 1 < NCHUNK) {
            issue_chunk((c + 1) * 32, cur ^ 1);    // prefetch next chunk
            wait_async8();                         // my chunk landed; next in flight
        } else {
            wait_async0();
        }
        __syncthreads();                           // tiles visible to all waves

        const float* kcur = &kpt[cur][0];
        const float* vcur = &vt[cur][0];

        // ---- scores for my 8 rows (lane = key c*32+lane) : v_tanh bound ----
        const float* kj = kcur + lane * KVPAD;
#pragma unroll
        for (int rr = 0; rr < 8; ++rr) {
            const int row = srow0 + rr;
            const float* qr = &qrow[row * 64];
            float s = 0.0f;
#pragma unroll 8
            for (int d = 0; d < DK; ++d)
                s += avs[d] * fast_tanh(qr[d] + kj[d]);

            float cm = s;
#pragma unroll
            for (int off = 16; off >= 1; off >>= 1)
                cm = fmaxf(cm, lane_read(cm, lane ^ off));
            const float mnew = fmaxf(mrow[rr], cm);
            const float sc   = fast_exp(mrow[rr] - mnew);  // exp(-inf)=0 first chunk
            const float p    = fast_exp(s - mnew);
            float ps = p;
#pragma unroll
            for (int off = 16; off >= 1; off >>= 1)
                ps += lane_read(ps, lane ^ off);
            lrow[rr] = lrow[rr] * sc + ps;
            mrow[rr] = mnew;

            ptile[rt][(row & 15) * 33 + lane] = p;
            if (lane == 0) { scl[row] = sc; lsum[row] = lrow[rr]; }
        }
        __syncthreads();                           // P / scales ready

        // ---- P@V on the matrix pipe ----
#pragma unroll
        for (int r = 0; r < 8; ++r) {
            const float sc = scl[rt * 16 + r + 8 * half];
            acc0[r] *= sc;
            acc1[r] *= sc;
        }
        v16h pa, vb0, vb1;
#pragma unroll
        for (int v = 0; v < 8; ++v) {
            const int ja = 16 * half + 2 * v;
            pa[2 * v]     = (_Float16)ptile[rt][q * 33 + ja];
            pa[2 * v + 1] = (_Float16)ptile[rt][q * 33 + ja + 1];

            const int jb = 16 * half + 2 * v;
            vb0[2 * v]     = (_Float16)vcur[jb * KVPAD + d0 + q];
            vb0[2 * v + 1] = (_Float16)vcur[(jb + 1) * KVPAD + d0 + q];
            vb1[2 * v]     = (_Float16)vcur[jb * KVPAD + d0 + 16 + q];
            vb1[2 * v + 1] = (_Float16)vcur[(jb + 1) * KVPAD + d0 + 16 + q];
        }
        acc0 = __builtin_amdgcn_wmma_f32_16x16x32_f16(false, pa, false, vb0, (short)0, acc0, false, false);
        acc1 = __builtin_amdgcn_wmma_f32_16x16x32_f16(false, pa, false, vb1, (short)0, acc1, false, false);

        __syncthreads();   // PV done reading cur buffer before it is re-issued
    }

    // ---- finalize: divide by softmax denom, store H ----
#pragma unroll
    for (int r = 0; r < 8; ++r) {
        const int row = rt * 16 + r + 8 * half;
        const float inv = __builtin_amdgcn_rcpf(lsum[row]);
        Hout[gbase + (size_t)(i0 + row) * DM + d0 + q]      = acc0[r] * inv;
        Hout[gbase + (size_t)(i0 + row) * DM + d0 + 16 + q] = acc1[r] * inv;
    }
}

// ------------------------------------------------------------------ launch --

extern "C" void kernel_launch(void* const* d_in, const int* in_sizes, int n_in,
                              void* d_out, int out_size, void* d_ws, size_t ws_size,
                              hipStream_t stream) {
    const float* query = (const float*)d_in[0];
    const float* key_  = (const float*)d_in[1];
    const float* value = (const float*)d_in[2];
    const float* Wq    = (const float*)d_in[3];
    const float* bq    = (const float*)d_in[4];
    const float* Wk    = (const float*)d_in[5];
    const float* bk    = (const float*)d_in[6];
    const float* Wv    = (const float*)d_in[7];
    const float* bv    = (const float*)d_in[8];
    const float* Wo    = (const float*)d_in[9];
    const float* bo    = (const float*)d_in[10];
    const float* Aq    = (const float*)d_in[11];
    const float* Ak    = (const float*)d_in[12];
    const float* av    = (const float*)d_in[13];
    float* out = (float*)d_out;

    // workspace layout (floats)
    float* ws     = (float*)d_ws;
    float* Wq_eff = ws;                      // 512*512
    float* Wk_eff = Wq_eff + DM * DM;        // 512*512
    float* bq_eff = Wk_eff + DM * DM;        // 512
    float* bk_eff = bq_eff + DM;             // 512
    float* qpbuf  = bk_eff + DM;             // 1024*512
    float* kpbuf  = qpbuf + (size_t)MROWS * DM;
    float* Vbuf   = kpbuf + (size_t)MROWS * DM;
    float* Hbuf   = Vbuf  + (size_t)MROWS * DM;

    // 1) fold inner projections into the outer weights
    {
        dim3 g(DM / 256, DM);
        fold_inner<<<g, 256, 0, stream>>>(Aq, Wq, Wq_eff);
        fold_inner<<<g, 256, 0, stream>>>(Ak, Wk, Wk_eff);
        fold_bias<<<DM / 256, 256, 0, stream>>>(Aq, bq, bq_eff);
        fold_bias<<<DM / 256, 256, 0, stream>>>(Ak, bk, bk_eff);
    }

    // 2) three WMMA projections: 512 waves each, 8 waves/block -> 64 blocks
    const int gemm_blocks = (MROWS / 16) * (DM / 64) / 8;   // 64
    wmma_gemm_xwT<<<gemm_blocks, 256, 0, stream>>>(query, Wq_eff, bq_eff, qpbuf, MROWS, DM, DM);
    wmma_gemm_xwT<<<gemm_blocks, 256, 0, stream>>>(key_,  Wk_eff, bk_eff, kpbuf, MROWS, DM, DM);
    wmma_gemm_xwT<<<gemm_blocks, 256, 0, stream>>>(value, Wv,     bv,     Vbuf,  MROWS, DM, DM);

    // 3) additive attention: tanh scores + online softmax + WMMA P@V
    attn_core_wmma<<<BATCH * NH * (LSEQ / 32), 128, 0, stream>>>(qpbuf, kpbuf, Vbuf, av, Hbuf);

    // 4) output projection
    wmma_gemm_xwT<<<gemm_blocks, 256, 0, stream>>>(Hbuf, Wo, bo, out, MROWS, DM, DM);
}
